// RBM_44848048504949
// MI455X (gfx1250) — compile-verified, matrix operand order checked by reading, not run on hardware
//
#include <hip/hip_runtime.h>

// ---------------------------------------------------------------------------
// RBM CD-k gradients on gfx1250 (MI455X), wave32 + v_wmma_f32_16x16x32_f16
// + GLOBAL_LOAD_ASYNC_TO_LDS_B128 staging (ASYNCcnt) + global_prefetch.
// B=16384, NV=4096, NH(A/P)=32. Memory-bound; f16 data / f32 WMMA accum.
// Gibbs-chain v lives in a 128MB f16 workspace -> resident in 192MB L2.
// ---------------------------------------------------------------------------

#define BNUM 16384
#define NV   4096
#define NH   32

typedef __attribute__((ext_vector_type(16))) _Float16 v16h;
typedef __attribute__((ext_vector_type(8)))  float    v8f;

// ---- workspace layout (in _Float16 elements) ------------------------------
#define WS_WA   ((size_t)0)                       // W_amp   f16 [NH][NV]
#define WS_WP   ((size_t)131072)                  // W_phase f16 [NH][NV]
#define WS_WTA  ((size_t)262144)                  // W_amp^T f16 [NV][NH]
#define WS_PH0A ((size_t)393216)                  // ph0 amp   f16 [B][NH]
#define WS_PH0P ((size_t)917504)                  // ph0 phase f16 [B][NH]
#define WS_PH2  ((size_t)1441792)                 // ph_k amp  f16 [B][NH]
#define WS_V    ((size_t)1966080)                 // v chain   f16 [B][NV]

// ---- output layout (floats) ----------------------------------------------
#define O_GWA   ((size_t)0)
#define O_GVBA  ((size_t)131072)
#define O_GHBA  ((size_t)135168)
#define O_GWP   ((size_t)135200)
#define O_GVBP  ((size_t)266272)
#define O_GHBP  ((size_t)270368)
#define O_TOTAL 270400

// ---- CDNA5 async global->LDS copy (16B per lane), guarded -----------------
#if __has_builtin(__builtin_amdgcn_global_load_async_to_lds_b128) && \
    __has_builtin(__builtin_amdgcn_s_wait_asynccnt)
#define HAVE_ASYNC_LDS 1
#else
#define HAVE_ASYNC_LDS 0
#endif

typedef int async_v4i __attribute__((vector_size(16)));

static __device__ __forceinline__ void async_copy16(void* dst_lds,
                                                    const void* src_global) {
#if HAVE_ASYNC_LDS
  async_v4i* s = (async_v4i*)(void*)src_global;
  async_v4i* d = (async_v4i*)dst_lds;
  __builtin_amdgcn_global_load_async_to_lds_b128(
      (__attribute__((address_space(1))) async_v4i*)s,
      (__attribute__((address_space(3))) async_v4i*)d, 0, 0);
#else
  *(int4*)dst_lds = *(const int4*)src_global;
#endif
}

static __device__ __forceinline__ void async_wait0() {
#if HAVE_ASYNC_LDS
  __builtin_amdgcn_s_wait_asynccnt(0);
#endif
}

static __device__ __forceinline__ float sigmoidf_(float x) {
  return 1.0f / (1.0f + __expf(-x));
}

// WMMA fragment load, 16-bit A/B layout:
// lane L: row = row16 + (L&15); kbase = (L>>4)*8;
// halves 0..7  -> K = kbase+0..7 ; halves 8..15 -> K = kbase+16..23
static __device__ __forceinline__ v16h load_frag(const _Float16* m, int row16,
                                                 int ld, int k0) {
  int lane = threadIdx.x & 31;
  const _Float16* p = m + (size_t)(row16 + (lane & 15)) * (size_t)ld + k0 +
                      ((lane >> 4) << 3);
  v16h r;
#pragma unroll
  for (int i = 0; i < 8; ++i) { r[i] = p[i]; r[i + 8] = p[16 + i]; }
  return r;
}

// Same pattern but source is f32 and caller pre-resolved the row pointer.
static __device__ __forceinline__ v16h load_frag_f32row(const float* rowbase,
                                                        int k0) {
  int lane = threadIdx.x & 31;
  const float* p = rowbase + k0 + ((lane >> 4) << 3);
  v16h r;
#pragma unroll
  for (int i = 0; i < 8; ++i) {
    r[i]     = (_Float16)p[i];
    r[i + 8] = (_Float16)p[16 + i];
  }
  return r;
}

static __device__ __forceinline__ v8f wmma16(v16h a, v16h b, v8f c) {
  return __builtin_amdgcn_wmma_f32_16x16x32_f16(false, a, false, b, (short)0,
                                                c, false, false);
}

static __device__ __forceinline__ v8f zero8() {
  v8f z;
#pragma unroll
  for (int i = 0; i < 8; ++i) z[i] = 0.0f;
  return z;
}

// ---------------------------------------------------------------------------
// k_prep: f16 weight copies (+ transpose of W_amp) and zero the output
// accumulators (required every launch: output is built with atomics).
// ---------------------------------------------------------------------------
__global__ __launch_bounds__(256) void k_prep(const float* __restrict__ Wa,
                                              const float* __restrict__ Wp,
                                              _Float16* __restrict__ ws,
                                              float* __restrict__ out) {
  int i = blockIdx.x * 256 + threadIdx.x;
  if (i < NH * NV) {
    float wa = Wa[i], wp = Wp[i];
    ws[WS_WA + i] = (_Float16)wa;
    ws[WS_WP + i] = (_Float16)wp;
    int h = i / NV, n = i % NV;
    ws[WS_WTA + (size_t)n * NH + h] = (_Float16)wa;
  }
  if (i < O_TOTAL) out[i] = 0.0f;
}

// ---------------------------------------------------------------------------
// k_h0: ph0_amp / ph0_phase = sigmoid(v0 @ W^T + hb), 16 batch rows per block.
// 8 waves split K=4096 (512 each); shared A fragment feeds 4 accumulators
// (amp/phase x two 16-wide N tiles). LDS reduce -> sigmoid -> f16 ph0 +
// atomic -mean(ph0) into ghb. Prefetch next v0 K-step (global_prefetch).
// ---------------------------------------------------------------------------
__global__ __launch_bounds__(256) void k_h0(const float* __restrict__ v0,
                                            const float* __restrict__ hba,
                                            const float* __restrict__ hbp,
                                            _Float16* __restrict__ ws,
                                            float* __restrict__ out) {
  __shared__ float red[8 * 1024];
  const _Float16* Wa = ws + WS_WA;
  const _Float16* Wp = ws + WS_WP;
  const int tid = threadIdx.x, w = tid >> 5, lane = tid & 31;
  const int row0 = blockIdx.x * 16;
  const float invb = 1.0f / (float)BNUM;

  const float* vrow = v0 + (size_t)(row0 + (lane & 15)) * NV;
  v8f acc[4];
#pragma unroll
  for (int q = 0; q < 4; ++q) acc[q] = zero8();

  for (int ks = 0; ks < 16; ++ks) {
    int k0 = w * 512 + ks * 32;
    if (ks + 1 < 16) __builtin_prefetch(vrow + k0 + 32, 0, 0);
    v16h a = load_frag_f32row(vrow, k0);
    acc[0] = wmma16(a, load_frag(Wa, 0,  NV, k0), acc[0]);
    acc[1] = wmma16(a, load_frag(Wa, 16, NV, k0), acc[1]);
    acc[2] = wmma16(a, load_frag(Wp, 0,  NV, k0), acc[2]);
    acc[3] = wmma16(a, load_frag(Wp, 16, NV, k0), acc[3]);
  }
#pragma unroll
  for (int q = 0; q < 4; ++q)
#pragma unroll
    for (int r = 0; r < 8; ++r)
      red[w * 1024 + q * 256 + lane * 8 + r] = acc[q][r];
  __syncthreads();

#pragma unroll
  for (int it = 0; it < 4; ++it) {
    int f = it * 256 + tid;
    float s = 0.0f;
#pragma unroll
    for (int wv = 0; wv < 8; ++wv) s += red[wv * 1024 + f];
    int q = f >> 8, lf = (f >> 3) & 31, r = f & 7;
    int row = r + ((lf >> 4) << 3);
    int hh  = ((q & 1) << 4) + (lf & 15);
    int net = q >> 1;  // 0 = amp, 1 = phase
    s += net ? hbp[hh] : hba[hh];
    float p = sigmoidf_(s);
    _Float16* dst = ws + (net ? WS_PH0P : WS_PH0A);
    dst[(size_t)(row0 + row) * NH + hh] = (_Float16)p;
    atomicAdd(out + (net ? O_GHBP : O_GHBA) + hh, -invb * p);
  }
}

// ---------------------------------------------------------------------------
// k_gibbs: per 16-row tile, k mean-field Gibbs steps (amp network).
//   v-step : v = sigmoid(ph @ W + vb)   -- K=32, one WMMA per 16-col tile;
//            v written f16 to L2-resident workspace (each wave owns the
//            column slice it will read back in the ph-step).
//   ph-step: ph = sigmoid(v @ W^T + hb) -- K=4096 split over 8 waves + LDS
//            reduction.
// ph0 tile is staged into LDS with async global->LDS copies.
// Last iteration emits ph2 (f16) plus +mean(v2)->gvb_amp, +mean(ph2)->ghb_amp.
// ---------------------------------------------------------------------------
__global__ __launch_bounds__(256) void k_gibbs(const float* __restrict__ vb,
                                               const float* __restrict__ hb,
                                               const int* __restrict__ kptr,
                                               _Float16* __restrict__ ws,
                                               float* __restrict__ out) {
  __shared__ float red[8 * 512];
  __shared__ __align__(16) _Float16 ph_lds[16 * NH];
  const _Float16* Wa  = ws + WS_WA;
  const _Float16* WTa = ws + WS_WTA;
  _Float16* vws = ws + WS_V;
  _Float16* ph2 = ws + WS_PH2;
  const int tid = threadIdx.x, w = tid >> 5, lane = tid & 31;
  const int row0 = blockIdx.x * 16;
  const float invb = 1.0f / (float)BNUM;
  int kk = kptr[0];
  if (kk < 1) kk = 1;

  // stage ph0 tile (16x32 f16 = 64 x 16B segments), async global->LDS
  if (tid < 64)
    async_copy16(&ph_lds[tid * 8], &ws[WS_PH0A + (size_t)row0 * NH + tid * 8]);
  async_wait0();
  __syncthreads();

  for (int g = 0; g < kk; ++g) {
    const bool last = (g == kk - 1);

    // ---- v-step: v = sigmoid(ph @ W + vb), wave w owns cols [w*512,w*512+512)
    v16h a = load_frag(ph_lds, 0, NH, 0);
    for (int i = 0; i < 32; ++i) {
      int n0 = (w * 32 + i) << 4;
      v16h b = load_frag(WTa, n0, NH, 0);
      v8f d = wmma16(a, b, zero8());
      int col = n0 + (lane & 15);
      float bias = vb[col];
      float csum = 0.0f;
#pragma unroll
      for (int r = 0; r < 8; ++r) {
        float p = sigmoidf_(d[r] + bias);
        vws[(size_t)(row0 + r + ((lane >> 4) << 3)) * NV + col] = (_Float16)p;
        csum += p;
      }
      if (last) atomicAdd(out + O_GVBA + col, invb * csum);
    }
    __threadfence();
    __syncthreads();

    // ---- ph-step: ph = sigmoid(v @ W^T + hb), split-K over 8 waves
    v8f acc[2];
    acc[0] = zero8();
    acc[1] = zero8();
    for (int ks = 0; ks < 16; ++ks) {
      int k0 = w * 512 + ks * 32;
      v16h av = load_frag(vws + (size_t)row0 * NV, 0, NV, k0);
      acc[0] = wmma16(av, load_frag(Wa, 0,  NV, k0), acc[0]);
      acc[1] = wmma16(av, load_frag(Wa, 16, NV, k0), acc[1]);
    }
#pragma unroll
    for (int nt = 0; nt < 2; ++nt)
#pragma unroll
      for (int r = 0; r < 8; ++r)
        red[w * 512 + nt * 256 + lane * 8 + r] = acc[nt][r];
    __syncthreads();

#pragma unroll
    for (int it = 0; it < 2; ++it) {
      int f = it * 256 + tid;
      float s = 0.0f;
#pragma unroll
      for (int wv = 0; wv < 8; ++wv) s += red[wv * 512 + f];
      int nt = f >> 8, lf = (f >> 3) & 31, r = f & 7;
      int row = r + ((lf >> 4) << 3);
      int hh  = (nt << 4) + (lf & 15);
      s += hb[hh];
      float p = sigmoidf_(s);
      ph_lds[row * NH + hh] = (_Float16)p;
      if (last) {
        ph2[(size_t)(row0 + row) * NH + hh] = (_Float16)p;
        atomicAdd(out + O_GHBA + hh, invb * p);
      }
    }
    __syncthreads();
  }
}

// ---------------------------------------------------------------------------
// k_op: split-K outer products
//   gW_amp   += inv_b * (ph2^T v2 - ph0^T v0)   (shared accumulator, ph0
//                                                fragment pre-negated)
//   gW_phase += inv_b * (-ph0p^T v0)
// Grid: x = 32 column chunks of 128, y = 16 batch chunks of 1024.
// f16 sources (v2, ph2, ph0, ph0p) stream in via async global->LDS b128
// copies into row-major scratch, then a cheap LDS->LDS transpose builds the
// K-contiguous fragment layouts. v0 (f32) is converted+transposed manually
// with a prefetch of the next chunk. Also harvests -mean(v0).
// ---------------------------------------------------------------------------
__global__ __launch_bounds__(256) void k_op(const float* __restrict__ v0,
                                            _Float16* __restrict__ ws,
                                            float* __restrict__ out) {
  __shared__ __align__(16) _Float16 vt2[128 * 40];
  __shared__ __align__(16) _Float16 vt0[128 * 40];
  __shared__ __align__(16) _Float16 pt2[32 * 40];
  __shared__ __align__(16) _Float16 pt0n[32 * 40];
  __shared__ __align__(16) _Float16 ptpn[32 * 40];
  __shared__ __align__(16) _Float16 s_v2[32 * 128];     // row-major scratch
  __shared__ __align__(16) _Float16 s_ph[3 * 32 * 32];  // p2 | p0a | p0p
  const _Float16* v2  = ws + WS_V;
  const _Float16* p2  = ws + WS_PH2;
  const _Float16* p0a = ws + WS_PH0A;
  const _Float16* p0p = ws + WS_PH0P;
  const int tid = threadIdx.x, w = tid >> 5, lane = tid & 31;
  const int nbase = blockIdx.x * 128;
  const int kbase = blockIdx.y * 1024;
  const float invb = 1.0f / (float)BNUM;

  v8f accA[2], accP[2];
#pragma unroll
  for (int mt = 0; mt < 2; ++mt) { accA[mt] = zero8(); accP[mt] = zero8(); }

  for (int step = 0; step < 32; ++step) {
    int kr = kbase + step * 32;
    __syncthreads();  // prior readers of vt*/pt*/s_* done

    // async: v2 chunk 32x128 f16 (512 x 16B) -> s_v2 row-major
#pragma unroll
    for (int it = 0; it < 2; ++it) {
      int seg = it * 256 + tid;
      int r = seg >> 4, c8 = (seg & 15) << 3;
      async_copy16(&s_v2[r * 128 + c8],
                   &v2[(size_t)(kr + r) * NV + nbase + c8]);
    }
    // async: three contiguous 32x32 f16 ph chunks (128 x 16B each)
    {
      int seg = tid;  // 256 threads cover segs 0..383 in 2 passes
      for (int pass = 0; pass < 2; ++pass, seg += 256) {
        if (seg < 384) {
          int which = seg >> 7, s2 = seg & 127;
          const _Float16* src = (which == 0) ? p2 : (which == 1 ? p0a : p0p);
          async_copy16(&s_ph[which * 1024 + s2 * 8],
                       &src[(size_t)kr * NH + s2 * 8]);
        }
      }
    }
    // manual: v0 chunk (f32 -> f16, transposed directly into vt0)
    for (int idx = tid; idx < 32 * 128; idx += 256) {
      int r = idx >> 7, c = idx & 127;
      vt0[c * 40 + r] = (_Float16)v0[(size_t)(kr + r) * NV + nbase + c];
    }
    if (step + 1 < 32) {  // prefetch next v0 chunk (32 rows x 512B)
      int pr = tid >> 3, pc = (tid & 7) << 4;
      __builtin_prefetch(&v0[(size_t)(kr + 32 + pr) * NV + nbase + pc], 0, 0);
    }
    async_wait0();
    __syncthreads();

    // LDS->LDS transpose into fragment-friendly (K-contiguous) layout
    for (int idx = tid; idx < 32 * 128; idx += 256) {
      int r = idx >> 7, c = idx & 127;
      vt2[c * 40 + r] = s_v2[r * 128 + c];
    }
    for (int idx = tid; idx < 32 * 32; idx += 256) {
      int r = idx >> 5, m = idx & 31;
      pt2[m * 40 + r]  = s_ph[r * 32 + m];
      pt0n[m * 40 + r] = (_Float16)(-(float)s_ph[1024 + r * 32 + m]);
      ptpn[m * 40 + r] = (_Float16)(-(float)s_ph[2048 + r * 32 + m]);
    }
    if (tid < 128) {  // -mean(v0) contributions while the chunk is staged
      float s = 0.0f;
#pragma unroll
      for (int r = 0; r < 32; ++r) s += (float)vt0[tid * 40 + r];
      atomicAdd(out + O_GVBA + nbase + tid, -invb * s);
      atomicAdd(out + O_GVBP + nbase + tid, -invb * s);
    }
    __syncthreads();

    v16h b2 = load_frag(vt2, w * 16, 40, 0);
    v16h b0 = load_frag(vt0, w * 16, 40, 0);
#pragma unroll
    for (int mt = 0; mt < 2; ++mt) {
      v16h a2 = load_frag(pt2,  mt * 16, 40, 0);
      v16h a0 = load_frag(pt0n, mt * 16, 40, 0);
      v16h ap = load_frag(ptpn, mt * 16, 40, 0);
      accA[mt] = wmma16(a2, b2, accA[mt]);
      accA[mt] = wmma16(a0, b0, accA[mt]);
      accP[mt] = wmma16(ap, b0, accP[mt]);
    }
  }

#pragma unroll
  for (int mt = 0; mt < 2; ++mt)
#pragma unroll
    for (int r = 0; r < 8; ++r) {
      int m   = mt * 16 + r + ((lane >> 4) << 3);
      int col = nbase + w * 16 + (lane & 15);
      atomicAdd(out + O_GWA + (size_t)m * NV + col, invb * accA[mt][r]);
      atomicAdd(out + O_GWP + (size_t)m * NV + col, invb * accP[mt][r]);
    }
}

// ---------------------------------------------------------------------------
extern "C" void kernel_launch(void* const* d_in, const int* in_sizes, int n_in,
                              void* d_out, int out_size, void* d_ws,
                              size_t ws_size, hipStream_t stream) {
  (void)in_sizes; (void)n_in; (void)out_size; (void)ws_size;
  const float* v0  = (const float*)d_in[0];
  const float* Wa  = (const float*)d_in[1];
  const float* vba = (const float*)d_in[2];
  const float* hba = (const float*)d_in[3];
  const float* Wp  = (const float*)d_in[4];
  // d_in[5] (vb_phase) is not used by the reference math (positive phase only)
  const float* hbp = (const float*)d_in[6];
  const int*   kp  = (const int*)d_in[7];
  float*    out = (float*)d_out;
  _Float16* ws  = (_Float16*)d_ws;

  k_prep <<<(O_TOTAL + 255) / 256, 256, 0, stream>>>(Wa, Wp, ws, out);
  k_h0   <<<BNUM / 16, 256, 0, stream>>>(v0, hba, hbp, ws, out);
  k_gibbs<<<BNUM / 16, 256, 0, stream>>>(vba, hba, kp, ws, out);
  k_op   <<<dim3(NV / 128, BNUM / 1024), 256, 0, stream>>>(v0, ws, out);
}
